// SparseCausal2DNA_22668837388653
// MI455X (gfx1250) — matrix operand order repaired and checked by reading, not compiled, not run on hardware
//
#include <hip/hip_runtime.h>

// ---------------------------------------------------------------------------
// MI455X (gfx1250): bf16 WMMA (v_wmma_f32_16x16x32_bf16) + Tensor Data Mover
// (tensor_load_to_lds / s_wait_tensorcnt) double-buffered GEMM pipeline.
// Problem is compute-bound (~143 GFLOP vs ~0.4GB HBM -> ~350 FLOP/B at
// 23.3 TB/s), so everything rides the bf16 matrix path; activations/weights
// are pre-converted once so GEMM tiles are raw TDM byte moves (zero VALU in
// the staging path). k/v window reuse in attention is served by the 192MB L2.
// ---------------------------------------------------------------------------

typedef __bf16 bf16;
typedef __attribute__((ext_vector_type(16))) __bf16 v16bf;
typedef __attribute__((ext_vector_type(8)))  float  v8f;
typedef __attribute__((ext_vector_type(4)))  unsigned int v4u;
typedef __attribute__((ext_vector_type(4)))  unsigned int v4ui;
typedef __attribute__((ext_vector_type(8)))  int v8i;
typedef __attribute__((ext_vector_type(4)))  int v4i;

__device__ __forceinline__ v8f wmma_bf16(v16bf a, v16bf b, v8f c) {
  return __builtin_amdgcn_wmma_f32_16x16x32_bf16(
      false, a, false, b, (short)0, c, false, false);
}

union Frag16 { v16bf v; v4ui u[2]; bf16 h[16]; };

__device__ __forceinline__ v8f vzero8() {
  v8f z = {0.f, 0.f, 0.f, 0.f, 0.f, 0.f, 0.f, 0.f};
  return z;
}

// Problem constants
#define BB    4
#define NN    4097
#define DIM   1024
#define HEADS 16
#define DHEAD 64
#define XBLK  256          // 4096/16 query blocks
#define MTOT  (BB * NN)    // 16388 flattened rows
#define NQKV  (3 * DIM)    // 3072

// ---------------------------------------------------------------------------
// TDM: 2D tile load Global->LDS (D# per CDNA5 ISA).
// data_size=2B, count=1, type=2; tensor_dim = remaining extent from the tile
// start so rows past the end of A are zero-filled by the TDM OOB rule.
// NOTE: the builtin only consumes integer descriptor words, so the compiler
// cannot see the LDS write -- consumers must call tdm_lds_acquire(lds_ptr)
// before reading the tile.
// ---------------------------------------------------------------------------
__device__ __forceinline__ void tdm_load_2d(
    unsigned lds_off, const void* gaddr,
    unsigned tensor_d0, unsigned tensor_d1,
    unsigned tile_d0, unsigned tile_d1, unsigned stride0)
{
  unsigned long long ga = (unsigned long long)(size_t)gaddr;
  v4u g0;
  g0[0] = 1u;                                            // count=1 (valid)
  g0[1] = lds_off;                                       // lds_addr
  g0[2] = (unsigned)(ga & 0xFFFFFFFFull);                // global_addr lo
  g0[3] = (unsigned)((ga >> 32) & 0x01FFFFFFull) | (2u << 30);  // hi | type=2
  v8i g1;
  g1[0] = (int)(1u << 16);                               // data_size = 2 bytes
  g1[1] = (int)((tensor_d0 & 0xFFFFu) << 16);            // tensor_dim0 lo
  g1[2] = (int)((tensor_d0 >> 16) | ((tensor_d1 & 0xFFFFu) << 16));
  g1[3] = (int)((tensor_d1 >> 16) | (tile_d0 << 16));    // tile_dim0
  g1[4] = (int)(tile_d1 & 0xFFFFu);                      // tile_dim1 (dim2=0)
  g1[5] = (int)stride0;                                  // dim0 stride lo
  g1[6] = 0;                                             // stride hi, dim1 stride
  g1[7] = 0;
  v4i gz = {0, 0, 0, 0};
#if defined(__clang_major__) && __clang_major__ >= 23
  v8i gz8 = {0, 0, 0, 0, 0, 0, 0, 0};
  __builtin_amdgcn_tensor_load_to_lds(g0, g1, gz, gz, gz8, 0);
#else
  __builtin_amdgcn_tensor_load_to_lds(g0, g1, gz, gz, 0);
#endif
}

// Compiler-visibility fence for TDM-written LDS. Passing the LDS pointer as an
// INPUT operand leaks its address into the asm, so "never-stored -> undef"
// global reasoning is defeated and the memory clobber blocks CSE/hoisting of
// the tile loads across iterations. Loads keep using the original
// __shared__-derived pointer (NOT an asm output), so addrspace(3) inference
// survives and the fragment loads stay plain ds_load_b128 (no volatile, no
// SCOPE_SYS serialization).
__device__ __forceinline__ void tdm_lds_acquire(const void* p) {
  asm volatile("" : : "r"(p) : "memory");
}

// window index j in [0,96): 0 -> BOS; [1,80] window; [81,96) padding (masked)
__device__ __forceinline__ int tok_of_j(int xi, int j) {
  if (j == 0 || j > 80) return 0;
  int bk = xi - 2 + (j - 1) / 16;
  if (bk < 0 || bk >= XBLK) return 0;
  return 1 + bk * 16 + ((j - 1) & 15);
}
__device__ __forceinline__ bool masked_j(int xi, int j, int qpos) {
  if (j == 0) return false;
  if (j > 80) return true;
  int bk = xi - 2 + (j - 1) / 16;
  if (bk < 0 || bk >= XBLK) return true;
  int kpos = bk * 16 + ((j - 1) & 15);
  return qpos < kpos;
}

// ---------------------------------------------------------------------------
// Pre-pass: f32 -> bf16 (row-major copy)
// ---------------------------------------------------------------------------
__global__ __launch_bounds__(256) void cvt_bf16_kernel(
    const float* __restrict__ src, bf16* __restrict__ dst)
{
  size_t i = (size_t)blockIdx.x * 256 + threadIdx.x;   // one uint4 (8 bf16) each
  const float4* p = (const float4*)src + i * 2;
  float4 a = p[0], b = p[1];
  union { uint4 u; bf16 h[8]; } st;
  st.h[0] = (bf16)a.x; st.h[1] = (bf16)a.y; st.h[2] = (bf16)a.z; st.h[3] = (bf16)a.w;
  st.h[4] = (bf16)b.x; st.h[5] = (bf16)b.y; st.h[6] = (bf16)b.z; st.h[7] = (bf16)b.w;
  ((uint4*)dst)[i] = st.u;
}

// Pre-pass: dst[n][k] = bf16(src[k][n])  (32x32 LDS tiled transpose)
__global__ __launch_bounds__(256) void transpose_cvt_kernel(
    const float* __restrict__ src, bf16* __restrict__ dst, int K, int N)
{
  __shared__ float tile[32][33];
  const int kt = blockIdx.y * 32, nt = blockIdx.x * 32;
  const int tx = threadIdx.x & 31, ty = threadIdx.x >> 5;
  #pragma unroll
  for (int i = 0; i < 4; ++i)
    tile[ty + i * 8][tx] = src[(size_t)(kt + ty + i * 8) * N + nt + tx];
  __syncthreads();
  #pragma unroll
  for (int i = 0; i < 4; ++i)
    dst[(size_t)(nt + ty + i * 8) * K + kt + tx] = (bf16)tile[tx][ty + i * 8];
}

// ---------------------------------------------------------------------------
// GEMM core: C(64x128) = A(bf16, MxK row-major) * BT(bf16, [N][K])^T.
// TDM double-buffered staging: wave0 issues the next (A,B) tile pair, then
// s_wait_tensorcnt 2 (current pair landed), barrier, 4 WMMA per wave/k-step.
// LDS per buffer: As[64][32] @ +0, Bs[128][32] @ +4096 (both are already the
// WMMA-fragment-friendly [row][K] layout -> 2x ds_load_b128 per fragment).
// ---------------------------------------------------------------------------
#define BUF_BYTES 12288
__device__ __forceinline__ void gemm_block_tdm(
    const bf16* __restrict__ A, const bf16* __restrict__ BT,
    int Ktot, int Ntot, int m0, int n0, char* smem, v8f acc[4])
{
  const int tid  = threadIdx.x;
  const int lane = tid & 31, wid = tid >> 5;
  const int wm = wid >> 1, wn = wid & 1;
  const int hh = lane >> 4, l16 = lane & 15;

  acc[0] = vzero8(); acc[1] = vzero8(); acc[2] = vzero8(); acc[3] = vzero8();

  const int S = Ktot / 32;
  const unsigned remM = (unsigned)(MTOT - m0);   // OOB rows -> TDM zero-fill
  const unsigned remN = (unsigned)(Ntot - n0);

  if (wid == 0) {
    tdm_load_2d(0u,    A  + (size_t)m0 * Ktot, (unsigned)Ktot, remM, 32u, 64u,  (unsigned)Ktot);
    tdm_load_2d(4096u, BT + (size_t)n0 * Ktot, (unsigned)Ktot, remN, 32u, 128u, (unsigned)Ktot);
  }

  for (int s = 0; s < S; ++s) {
    const unsigned cbase = (s & 1) ? BUF_BYTES : 0u;
    if (wid == 0) {
      if (s + 1 < S) {
        const int kn = (s + 1) * 32;
        const unsigned nbase = ((s + 1) & 1) ? BUF_BYTES : 0u;
        tdm_load_2d(nbase,         A  + (size_t)m0 * Ktot + kn,
                    (unsigned)(Ktot - kn), remM, 32u, 64u,  (unsigned)Ktot);
        tdm_load_2d(nbase + 4096u, BT + (size_t)n0 * Ktot + kn,
                    (unsigned)(Ktot - kn), remN, 32u, 128u, (unsigned)Ktot);
        __builtin_amdgcn_s_wait_tensorcnt(2);   // current pair complete
      } else {
        __builtin_amdgcn_s_wait_tensorcnt(0);
      }
    }
    __syncthreads();
    tdm_lds_acquire(smem);   // TDM's LDS writes are now compiler-visible

    const bf16* As = (const bf16*)(smem + cbase);
    const bf16* Bs = (const bf16*)(smem + cbase + 4096);

    Frag16 fa;
    const bf16* ap = &As[(wm * 16 + l16) * 32];
    fa.u[0] = *(const v4ui*)(ap + 8 * hh);
    fa.u[1] = *(const v4ui*)(ap + 8 * hh + 16);
    #pragma unroll
    for (int nt = 0; nt < 4; ++nt) {
      Frag16 fb;
      const bf16* bp2 = &Bs[(wn * 64 + nt * 16 + l16) * 32];
      fb.u[0] = *(const v4ui*)(bp2 + 8 * hh);
      fb.u[1] = *(const v4ui*)(bp2 + 8 * hh + 16);
      acc[nt] = wmma_bf16(fa.v, fb.v, acc[nt]);
    }
    __syncthreads();
  }
}

// ---------------------------------------------------------------------------
// Kernel 1: qkv = xbf @ wqkvT^T; scatter q(scaled)/k/v head-major bf16;
// v at t==0 -> aobf row 0 (out_bos path).
// ---------------------------------------------------------------------------
__global__ __launch_bounds__(256) void qkv_kernel(
    const bf16* __restrict__ xbf, const bf16* __restrict__ wqkvT,
    bf16* __restrict__ qbf, bf16* __restrict__ kbf, bf16* __restrict__ vbf,
    bf16* __restrict__ aobf)
{
  __shared__ alignas(128) char smem[2 * BUF_BYTES];
  const int m0 = blockIdx.y * 64, n0 = blockIdx.x * 128;
  v8f acc[4];
  gemm_block_tdm(xbf, wqkvT, DIM, NQKV, m0, n0, smem, acc);

  const int lane = threadIdx.x & 31, wid = threadIdx.x >> 5;
  const int wm = wid >> 1, wn = wid & 1;
  const int hh = lane >> 4, l16 = lane & 15;
  const int mbase = m0 + wm * 16, nbase = n0 + wn * 64;

  #pragma unroll
  for (int nt = 0; nt < 4; ++nt) {
    #pragma unroll
    for (int r = 0; r < 8; ++r) {
      int gr = mbase + r + 8 * hh;
      if (gr >= MTOT) continue;
      int gc = nbase + nt * 16 + l16;
      float val = acc[nt][r];
      int b = gr / NN, t = gr % NN;
      int part = gc >> 10, c = gc & 1023;
      int h = c >> 6, dd = c & 63;
      if (part == 0) {
        if (t >= 1)
          qbf[(((size_t)(b * HEADS + h)) * 4096 + (t - 1)) * DHEAD + dd] =
              (bf16)(val * 0.125f);                   // d^-0.5
      } else if (part == 1) {
        kbf[(((size_t)(b * HEADS + h)) * NN + t) * DHEAD + dd] = (bf16)val;
      } else {
        vbf[(((size_t)(b * HEADS + h)) * NN + t) * DHEAD + dd] = (bf16)val;
        if (t == 0) aobf[(size_t)b * NN * DIM + c] = (bf16)val;
      }
    }
  }
}

// ---------------------------------------------------------------------------
// Kernel 2: windowed attention + talking heads. One WG per (b, block xi);
// 16 waves = 16 heads. 12 WMMA sim + 12 WMMA pv per wave; shuffle softmax;
// head mix via [i][j][h] LDS layout (contiguous b128 reads).
// ---------------------------------------------------------------------------
__global__ __launch_bounds__(512) void attn_kernel(
    const bf16* __restrict__ qbf, const bf16* __restrict__ kbf,
    const bf16* __restrict__ vbf, const float* __restrict__ wth,
    bf16* __restrict__ aobf)
{
  __shared__ bf16 attn_s[16 * 96 * 16];   // [i][j][h], 48KB

  const int xi = blockIdx.x, b = blockIdx.y;
  const int tid = threadIdx.x, lane = tid & 31, h = tid >> 5;
  const int hh = lane >> 4, l16 = lane & 15;

  const size_t qb = ((size_t)(b * HEADS + h)) * 4096 * DHEAD;
  const size_t kb = ((size_t)(b * HEADS + h)) * NN * DHEAD;

  // warm L0/L2 for the v window while we do qk + softmax
  __builtin_prefetch(vbf + kb + (size_t)tok_of_j(xi, lane) * DHEAD, 0, 0);
  __builtin_prefetch(vbf + kb + (size_t)tok_of_j(xi, lane + 48) * DHEAD, 0, 0);

  // q A-fragments (16x64 => 2x 16x32)
  Frag16 aq[2];
  {
    const bf16* qrow = qbf + qb + (size_t)(xi * 16 + l16) * DHEAD;
    #pragma unroll
    for (int kk = 0; kk < 2; ++kk) {
      aq[kk].u[0] = *(const v4ui*)(qrow + kk * 32 + 16 * hh);
      aq[kk].u[1] = *(const v4ui*)(qrow + kk * 32 + 16 * hh + 8);
    }
  }

  // sim = q.k^T : 6 key tiles x 2 k-steps
  float s[6][8];
  #pragma unroll
  for (int nt = 0; nt < 6; ++nt) {
    int j = nt * 16 + l16;
    int tok = tok_of_j(xi, j);
    const bf16* krow = kbf + kb + (size_t)tok * DHEAD + 16 * hh;
    Frag16 b0, b1;
    b0.u[0] = *(const v4ui*)(krow + 0);
    b0.u[1] = *(const v4ui*)(krow + 8);
    b1.u[0] = *(const v4ui*)(krow + 32);
    b1.u[1] = *(const v4ui*)(krow + 40);
    v8f c = vzero8();
    c = wmma_bf16(aq[0].v, b0.v, c);
    c = wmma_bf16(aq[1].v, b1.v, c);
    #pragma unroll
    for (int r = 0; r < 8; ++r) s[nt][r] = c[r];
  }

  // mask + softmax (row spread over 16 lanes x 6 tiles)
  #pragma unroll
  for (int r = 0; r < 8; ++r) {
    int qpos = xi * 16 + r + 8 * hh;
    float m = -3.402823466e38f;
    #pragma unroll
    for (int nt = 0; nt < 6; ++nt) {
      int j = nt * 16 + l16;
      if (masked_j(xi, j, qpos)) s[nt][r] = -3.402823466e38f;
      m = fmaxf(m, s[nt][r]);
    }
    m = fmaxf(m, __shfl_xor(m, 1, 32));
    m = fmaxf(m, __shfl_xor(m, 2, 32));
    m = fmaxf(m, __shfl_xor(m, 4, 32));
    m = fmaxf(m, __shfl_xor(m, 8, 32));
    float sum = 0.f;
    #pragma unroll
    for (int nt = 0; nt < 6; ++nt) { s[nt][r] = __expf(s[nt][r] - m); sum += s[nt][r]; }
    sum += __shfl_xor(sum, 1, 32);
    sum += __shfl_xor(sum, 2, 32);
    sum += __shfl_xor(sum, 4, 32);
    sum += __shfl_xor(sum, 8, 32);
    float inv = 1.f / sum;
    #pragma unroll
    for (int nt = 0; nt < 6; ++nt) s[nt][r] *= inv;
  }

  // publish probs
  #pragma unroll
  for (int nt = 0; nt < 6; ++nt) {
    int j = nt * 16 + l16;
    #pragma unroll
    for (int r = 0; r < 8; ++r)
      attn_s[((r + 8 * hh) * 96 + j) * 16 + h] = (bf16)s[nt][r];
  }
  __syncthreads();

  // talking heads: mixed[g] = sum_h w_th[g,h]*attn[h]; build WMMA A-frags
  float wt[16];
  #pragma unroll
  for (int t = 0; t < 16; ++t) wt[t] = wth[h * 16 + t];

  Frag16 am[3];
  #pragma unroll
  for (int kk2 = 0; kk2 < 3; ++kk2) {
    #pragma unroll
    for (int e = 0; e < 16; ++e) {
      int K = kk2 * 32 + 8 * hh + (e < 8 ? e : e + 8);
      union { v4ui u[2]; bf16 hv[16]; } t16;
      const v4ui* p = (const v4ui*)&attn_s[(l16 * 96 + K) * 16];
      t16.u[0] = p[0]; t16.u[1] = p[1];
      float accv = 0.f;
      #pragma unroll
      for (int t = 0; t < 16; ++t) accv += wt[t] * (float)t16.hv[t];
      am[kk2].h[e] = (bf16)accv;
    }
  }

  // out = mixed_attn(16x96) @ v(96x64) -> aobf (bf16, feeds final TDM GEMM)
  #pragma unroll
  for (int nd = 0; nd < 4; ++nd) {
    int d = nd * 16 + l16;
    v8f c = vzero8();
    #pragma unroll
    for (int kk2 = 0; kk2 < 3; ++kk2) {
      Frag16 fv;
      #pragma unroll
      for (int e = 0; e < 16; ++e) {
        int j = kk2 * 32 + 8 * hh + (e < 8 ? e : e + 8);
        fv.h[e] = vbf[kb + (size_t)tok_of_j(xi, j) * DHEAD + d];
      }
      c = wmma_bf16(am[kk2].v, fv.v, c);
    }
    #pragma unroll
    for (int r = 0; r < 8; ++r) {
      int t = 1 + xi * 16 + r + 8 * hh;
      aobf[((size_t)b * NN + t) * DIM + h * DHEAD + d] = (bf16)c[r];
    }
  }
}

// ---------------------------------------------------------------------------
// Kernel 3: out = aobf @ woutT^T + b_out (f32 result)
// ---------------------------------------------------------------------------
__global__ __launch_bounds__(256) void out_kernel(
    const bf16* __restrict__ aobf, const bf16* __restrict__ woutT,
    const float* __restrict__ bout, float* __restrict__ out)
{
  __shared__ alignas(128) char smem[2 * BUF_BYTES];
  const int m0 = blockIdx.y * 64, n0 = blockIdx.x * 128;
  v8f acc[4];
  gemm_block_tdm(aobf, woutT, DIM, DIM, m0, n0, smem, acc);

  const int lane = threadIdx.x & 31, wid = threadIdx.x >> 5;
  const int wm = wid >> 1, wn = wid & 1;
  const int hh = lane >> 4, l16 = lane & 15;
  const int mbase = m0 + wm * 16, nbase = n0 + wn * 64;

  #pragma unroll
  for (int nt = 0; nt < 4; ++nt) {
    #pragma unroll
    for (int r = 0; r < 8; ++r) {
      int gr = mbase + r + 8 * hh;
      if (gr >= MTOT) continue;
      int gc = nbase + nt * 16 + l16;
      out[(size_t)gr * DIM + gc] = acc[nt][r] + bout[gc];
    }
  }
}

// ---------------------------------------------------------------------------
// Workspace (bytes):
//   xbf   @ 0          33,562,624
//   wqkvT @ 33562624    6,291,456
//   woutT @ 39854080    2,097,152
//   qbf   @ 41951232   33,554,432
//   kbf   @ 75505664   33,562,624
//   vbf   @ 109068288  33,562,624
//   aobf  @ 142630912  33,562,624   (total ~168MB)
// ---------------------------------------------------------------------------
extern "C" void kernel_launch(void* const* d_in, const int* in_sizes, int n_in,
                              void* d_out, int out_size, void* d_ws, size_t ws_size,
                              hipStream_t stream) {
  const float* x    = (const float*)d_in[0];
  const float* wqkv = (const float*)d_in[1];
  const float* wout = (const float*)d_in[2];
  const float* bout = (const float*)d_in[3];
  const float* wth  = (const float*)d_in[4];
  float* out = (float*)d_out;

  char* ws = (char*)d_ws;
  bf16* xbf   = (bf16*)(ws);
  bf16* wqkvT = (bf16*)(ws + 33562624);
  bf16* woutT = (bf16*)(ws + 39854080);
  bf16* qbf   = (bf16*)(ws + 41951232);
  bf16* kbf   = (bf16*)(ws + 75505664);
  bf16* vbf   = (bf16*)(ws + 109068288);
  bf16* aobf  = (bf16*)(ws + 142630912);

  // pre-passes: one-time bf16 conversion / weight transposes
  cvt_bf16_kernel<<<dim3(MTOT * DIM / 8 / 256), 256, 0, stream>>>(x, xbf);
  transpose_cvt_kernel<<<dim3(NQKV / 32, DIM / 32), 256, 0, stream>>>(wqkv, wqkvT, DIM, NQKV);
  transpose_cvt_kernel<<<dim3(DIM / 32, DIM / 32), 256, 0, stream>>>(wout, woutT, DIM, DIM);

  // qkv projection: M=16388, N=3072
  qkv_kernel<<<dim3(24, 257), 256, 0, stream>>>(xbf, wqkvT, qbf, kbf, vbf, aobf);
  // attention: one WG per (block, batch), 16 waves = 16 heads
  attn_kernel<<<dim3(XBLK, BB), 512, 0, stream>>>(qbf, kbf, vbf, wth, aobf);
  // output projection: M=16388, N=1024
  out_kernel<<<dim3(8, 257), 256, 0, stream>>>(aobf, woutT, bout, out);
}
